// QuantizedLinear_30339648979186
// MI455X (gfx1250) — matrix operand order repaired and checked by reading, not compiled, not run on hardware
//
#include <hip/hip_runtime.h>
#include <stdint.h>

typedef __attribute__((ext_vector_type(16))) _Float16 v16h;
typedef __attribute__((ext_vector_type(8)))  _Float16 v8h;
typedef __attribute__((ext_vector_type(2)))  _Float16 v2h;
typedef __attribute__((ext_vector_type(8)))  float    v8f;

#define IN_F   4096
#define OUT_F  11008
#define GS     128
#define NG     (IN_F / GS)

#define BM 128
#define BN 128
#define KT 64
#define LDK (KT + 8)          // padded row (halves) to spread LDS banks
#define NSTEP (IN_F / KT)

__global__ __launch_bounds__(256)
void wq4_gemm_kernel(const float* __restrict__ X,
                     const int*   __restrict__ WP,
                     const float* __restrict__ SC,
                     const float* __restrict__ ZP,
                     const float* __restrict__ BIAS,
                     float* __restrict__ Y)
{
  __shared__ __align__(16) _Float16 lA[2][BM][LDK];
  __shared__ __align__(16) _Float16 lB[2][BN][LDK];

  const int tid  = threadIdx.x;
  const int lane = tid & 31;
  const int wav  = tid >> 5;
  const int waveM = wav >> 2;          // 0..1 : 64 rows each
  const int waveN = wav & 3;           // 0..3 : 32 cols each

  const int blockM = blockIdx.y * BM;
  const int blockN = blockIdx.x * BN;

  // X staging: each thread loads 8 consecutive floats per pass, 4 passes
  const int xc = (tid & 7) * 8;        // k offset within K-tile
  const int xr = tid >> 3;             // row 0..31 (+32 per pass)
  // W staging: 2 threads per output row; each handles 16 packed ints (32 k)
  const int wh = tid & 1;
  const int wn = tid >> 1;             // 0..127

  float4 xreg[4][2];
  int4   wreg[4];
  float  wscale = 0.f, wzero = 0.f;

  v8f acc[4][2];
  #pragma unroll
  for (int i = 0; i < 4; ++i)
    #pragma unroll
    for (int j = 0; j < 2; ++j)
      #pragma unroll
      for (int e = 0; e < 8; ++e)
        acc[i][j][e] = 0.0f;

  const float* xbase = X + (size_t)blockM * IN_F;
  const int*   wbase = WP + (size_t)(blockN + wn) * (IN_F / 2) + wh * 16;
  const float* scp   = SC + (size_t)(blockN + wn) * NG;
  const float* zpp   = ZP + (size_t)(blockN + wn) * NG;

  auto gload = [&](int kk) {
    #pragma unroll
    for (int p = 0; p < 4; ++p) {
      const float* q = xbase + (size_t)(xr + 32 * p) * IN_F + kk + xc;
      xreg[p][0] = *(const float4*)(q);
      xreg[p][1] = *(const float4*)(q + 4);
    }
    const int* w = wbase + (kk >> 1);
    #pragma unroll
    for (int p = 0; p < 4; ++p)
      wreg[p] = *(const int4*)(w + p * 4);
    const int g = kk / GS;               // K-step (64) stays inside one group (128)
    wscale = scp[g];
    wzero  = zpp[g];
  };

  auto stage = [&](int buf) {
    // ---- X: f32 -> f16, b128 stores ----
    #pragma unroll
    for (int p = 0; p < 4; ++p) {
      v8h h;
      h[0] = (_Float16)xreg[p][0].x; h[1] = (_Float16)xreg[p][0].y;
      h[2] = (_Float16)xreg[p][0].z; h[3] = (_Float16)xreg[p][0].w;
      h[4] = (_Float16)xreg[p][1].x; h[5] = (_Float16)xreg[p][1].y;
      h[6] = (_Float16)xreg[p][1].z; h[7] = (_Float16)xreg[p][1].w;
      *(v8h*)&lA[buf][xr + 32 * p][xc] = h;
    }
    // ---- W: nibble pair -> f16 via 0x6400 exponent trick + v_pk_fma ----
    const _Float16 hs = (_Float16)wscale;
    const _Float16 hz = (_Float16)(-wzero * wscale);
    const v2h s2 = {hs, hs};
    const v2h z2 = {hz, hz};
    const v2h k1024 = {(_Float16)1024.f, (_Float16)1024.f};
    #pragma unroll
    for (int p = 0; p < 4; ++p) {
      const uint32_t bb[4] = {(uint32_t)wreg[p].x, (uint32_t)wreg[p].y,
                              (uint32_t)wreg[p].z, (uint32_t)wreg[p].w};
      v8h h;
      #pragma unroll
      for (int j = 0; j < 4; ++j) {
        // low half = even k (hi nibble), high half = odd k (lo nibble)
        uint32_t t = ((bb[j] >> 4) & 0xFu) | ((bb[j] & 0xFu) << 16) | 0x64006400u;
        v2h q  = __builtin_bit_cast(v2h, t) - k1024;   // exact integer q in f16
        v2h w2 = q * s2 + z2;                          // pk_fma: q*s - z*s
        h[2 * j]     = w2[0];
        h[2 * j + 1] = w2[1];
      }
      *(v8h*)&lB[buf][wn][wh * 32 + p * 8] = h;
    }
  };

  const int lm  = lane & 15;
  const int lhi = lane >> 4;

  auto compute = [&](int buf) {
    #pragma unroll
    for (int kh = 0; kh < 2; ++kh) {
      v16h a[4], b[2];
      #pragma unroll
      for (int mt = 0; mt < 4; ++mt) {
        // A 16x32 f16: lanes 0-15 K{0..7,16..23}, lanes 16-31 +8
        const _Float16* p = &lA[buf][waveM * 64 + mt * 16 + lm][kh * 32 + lhi * 8];
        union { v16h v; v8h h[2]; } u;
        u.h[0] = *(const v8h*)(p);
        u.h[1] = *(const v8h*)(p + 16);
        a[mt] = u.v;
      }
      #pragma unroll
      for (int nt = 0; nt < 2; ++nt) {
        // B 32x16 f16: lanes 0-15 K{0..7,16..23}, lanes 16-31 +8
        const _Float16* p = &lB[buf][waveN * 32 + nt * 16 + lm][kh * 32 + lhi * 8];
        union { v16h v; v8h h[2]; } u;
        u.h[0] = *(const v8h*)(p);
        u.h[1] = *(const v8h*)(p + 16);
        b[nt] = u.v;
      }
      #pragma unroll
      for (int mt = 0; mt < 4; ++mt)
        #pragma unroll
        for (int nt = 0; nt < 2; ++nt)
          acc[mt][nt] = __builtin_amdgcn_wmma_f32_16x16x32_f16(
              false, a[mt], false, b[nt], (short)0, acc[mt][nt], false, false);
    }
  };

  gload(0);
  stage(0);
  __syncthreads();

  for (int s = 0; s < NSTEP; ++s) {
    if (s + 1 < NSTEP) gload((s + 1) * KT);   // issue next tile's global loads early
    compute(s & 1);                           // WMMAs on current LDS buffer
    if (s + 1 < NSTEP) stage((s + 1) & 1);    // convert/dequant into other buffer
    __syncthreads();                          // one barrier per K-step
  }

  // ---- epilogue: bias add + non-temporal store (write-once 360 MB) ----
  const int rowBase = blockM + waveM * 64 + 8 * lhi;
  const int colBase = blockN + waveN * 32 + lm;
  #pragma unroll
  for (int nt = 0; nt < 2; ++nt) {
    const float bv = BIAS[colBase + nt * 16];
    #pragma unroll
    for (int mt = 0; mt < 4; ++mt) {
      #pragma unroll
      for (int v = 0; v < 8; ++v) {
        const size_t off = (size_t)(rowBase + mt * 16 + v) * OUT_F + (colBase + nt * 16);
        __builtin_nontemporal_store(acc[mt][nt][v] + bv, Y + off);
      }
    }
  }
}

extern "C" void kernel_launch(void* const* d_in, const int* in_sizes, int n_in,
                              void* d_out, int out_size, void* d_ws, size_t ws_size,
                              hipStream_t stream) {
  (void)n_in; (void)out_size; (void)d_ws; (void)ws_size;
  const float* x    = (const float*)d_in[0];
  const int*   wp   = (const int*)d_in[1];
  const float* sc   = (const float*)d_in[2];
  const float* zp   = (const float*)d_in[3];
  const float* bias = (const float*)d_in[4];
  float* y = (float*)d_out;

  const int M = in_sizes[0] / IN_F;           // 4*2048 = 8192, divisible by BM
  dim3 grid(OUT_F / BN, M / BM);              // 86 x 64 blocks
  wq4_gemm_kernel<<<grid, dim3(256), 0, stream>>>(x, wp, sc, zp, bias, y);
}
